// CooccurrenceGraph_11991548691263
// MI455X (gfx1250) — matrix operand order repaired and checked by reading, not compiled, not run on hardware
//
#include <hip/hip_runtime.h>

typedef _Float16 half_t;
typedef __attribute__((ext_vector_type(16))) _Float16 v16h;
typedef __attribute__((ext_vector_type(8)))  float    v8f;

#define BDIM 256
#define NCLS 80
#define HDIM 256

// ---- LDS layout (element strides; byte offsets) ----
#define XS 264   // f16 row stride for X/Q/K/V/O tiles (256 + 8 pad)
#define SS 84    // f32 row stride for scores
#define AS 104   // f16 row stride for attn probs (96 cols + 8 pad)

#define X_OFF    0        // 80*264*2 = 42240
#define Q_OFF    42240
#define K_OFF    84480
#define V_OFF    126720
#define S_OFF    168960   // 80*84*4 = 26880
#define MASK_OFF 195840   // 80*4
#define SMEM_TOTAL 196160
#define A_OFF Q_OFF       // attn aliases Q (Q dead after scores + barrier)
#define O_OFF X_OFF       // attn@V output aliases X (X dead after phase 1)

// ---------------- WMMA fragment loaders (contiguous k order: k = 16*(lane/16) + element) ----------------
// A-matrix 16x32 f16: lanes 0-15 -> M=lane, VGPR j: K=2j,2j+1 ;
//                     lanes 16-31 -> same M, K=16+2j.
__device__ __forceinline__ v16h load_a_f16(const half_t* __restrict__ src, int stride) {
  const int lane = threadIdx.x & 31;
  const int r = lane & 15, hh = lane >> 4;
  v16h a;
#pragma unroll
  for (int j = 0; j < 8; ++j) {
    const int kk = hh * 16 + 2 * j;
    a[2 * j]     = src[r * stride + kk];
    a[2 * j + 1] = src[r * stride + kk + 1];
  }
  return a;
}

// B-matrix 32x16 f16 loaded from ROW-MAJOR [n][k] source (i.e. B = W^T implicitly):
// lanes 0-15 -> N=lane, VGPR j: K=2j,2j+1 ; lanes 16-31 -> same N, K=16+2j.
__device__ __forceinline__ v16h load_b_f16(const half_t* __restrict__ src, int stride) {
  const int lane = threadIdx.x & 31;
  const int r = lane & 15, hh = lane >> 4;
  v16h b;
#pragma unroll
  for (int j = 0; j < 8; ++j) {
    const int kk = hh * 16 + 2 * j;
    b[2 * j]     = src[r * stride + kk];
    b[2 * j + 1] = src[r * stride + kk + 1];
  }
  return b;
}

// Same as load_b_f16 but source is fp32 in global (weights; L2-resident), cvt on the fly.
__device__ __forceinline__ v16h load_b_f32g(const float* __restrict__ src, int stride) {
  const int lane = threadIdx.x & 31;
  const int r = lane & 15, hh = lane >> 4;
  v16h b;
#pragma unroll
  for (int j = 0; j < 8; ++j) {
    const int kk = hh * 16 + 2 * j;
    b[2 * j]     = (half_t)src[r * stride + kk];
    b[2 * j + 1] = (half_t)src[r * stride + kk + 1];
  }
  return b;
}

// B fragment where the logical B[k][n] comes from a row-major [k][n] LDS buffer
// (used for attn @ V: k = class index m, n = hidden h). Rows >= kmax read as 0.
__device__ __forceinline__ v16h load_b_colmaj(const half_t* __restrict__ src, int stride,
                                              int k_base, int n_base, int kmax) {
  const int lane = threadIdx.x & 31;
  const int r = lane & 15, hh = lane >> 4;
  v16h b;
#pragma unroll
  for (int j = 0; j < 8; ++j) {
    const int k0 = k_base + hh * 16 + 2 * j;
    b[2 * j]     = (k0     < kmax) ? src[(k0)     * stride + n_base + r] : (half_t)0.f;
    b[2 * j + 1] = (k0 + 1 < kmax) ? src[(k0 + 1) * stride + n_base + r] : (half_t)0.f;
  }
  return b;
}

__device__ __forceinline__ v8f wmma_step(v16h a, v16h b, v8f c) {
  return __builtin_amdgcn_wmma_f32_16x16x32_f16(false, a, false, b, (short)0, c, false, false);
}

// One projection: dst[n][o] = f16( sum_h Xs[n][h] * W[o][h] + bias[o] ), 80x256 tiles.
// Weight B-fragments for a given o-tile are hoisted into VGPRs once and reused
// across all 5 n-tiles (5x fewer global loads + f32->f16 conversions).
__device__ __forceinline__ void project(const half_t* __restrict__ Xs,
                                        const float* __restrict__ W,
                                        const float* __restrict__ bias,
                                        half_t* __restrict__ dst,
                                        int wave, int tid) {
  for (int ot = wave; ot < 16; ot += 8) {      // 16 o-tiles over 8 waves
    v16h bf[8];
#pragma unroll
    for (int kc = 0; kc < 8; ++kc)
      bf[kc] = load_b_f32g(W + (size_t)(ot * 16) * HDIM + kc * 32, HDIM);

    const int lane = tid & 31;
    const int ncol = lane & 15;
    const int mbase = (lane >> 4) * 8;
    const float bval = bias[ot * 16 + ncol];

#pragma unroll 1
    for (int nt = 0; nt < 5; ++nt) {           // reuse bf[] across n-tiles
      v8f acc = {};
#pragma unroll
      for (int kc = 0; kc < 8; ++kc) {
        v16h a = load_a_f16(Xs + nt * 16 * XS + kc * 32, XS);
        acc = wmma_step(a, bf[kc], acc);
      }
#pragma unroll
      for (int v = 0; v < 8; ++v)
        dst[(nt * 16 + mbase + v) * XS + ot * 16 + ncol] = (half_t)(acc[v] + bval);
    }
  }
}

__global__ __launch_bounds__(BDIM, 1)
void cooc_attn_kernel(const float* __restrict__ x,
                      const float* __restrict__ Wq, const float* __restrict__ bq,
                      const float* __restrict__ Wk, const float* __restrict__ bk,
                      const float* __restrict__ Wv, const float* __restrict__ bv,
                      const float* __restrict__ Wo, const float* __restrict__ bo,
                      const float* __restrict__ cooc, const int* __restrict__ labels,
                      float* __restrict__ out) {
  __shared__ alignas(16) unsigned char smem[SMEM_TOTAL];
  half_t* Xs = (half_t*)(smem + X_OFF);
  half_t* Qs = (half_t*)(smem + Q_OFF);
  half_t* Ks = (half_t*)(smem + K_OFF);
  half_t* Vs = (half_t*)(smem + V_OFF);
  half_t* As = (half_t*)(smem + A_OFF);
  half_t* Os = (half_t*)(smem + O_OFF);
  float*  Ss = (float*)(smem + S_OFF);
  float*  Ms = (float*)(smem + MASK_OFF);

  const int tid  = threadIdx.x;
  const int wave = tid >> 5;
  const int b    = blockIdx.x;
  const float* xb = x + (size_t)b * NCLS * HDIM;

  // L2 prefetch hints for the (shared, L2-resident) weight matrices.
  __builtin_prefetch(Wq + tid * 256, 0, 1);
  __builtin_prefetch(Wk + tid * 256, 0, 1);
  __builtin_prefetch(Wv + tid * 256, 0, 1);
  __builtin_prefetch(Wo + tid * 256, 0, 1);

  // Phase 0: stage x[b] into LDS as f16; build label mask.
  for (int idx = tid; idx < NCLS * HDIM; idx += BDIM) {
    const int r = idx >> 8, c = idx & 255;
    Xs[r * XS + c] = (half_t)xb[idx];
  }
  if (tid < NCLS) Ms[tid] = (float)labels[b * NCLS + tid] * 0.8f + 0.2f;
  __syncthreads();

  // Phase 1: Q/K/V projections (WMMA, fp32 accumulate, f16 to LDS).
  project(Xs, Wq, bq, Qs, wave, tid);
  project(Xs, Wk, bk, Ks, wave, tid);
  project(Xs, Wv, bv, Vs, wave, tid);
  __syncthreads();

  // Phase 2: scores[n][m] = (q.k)/16 * cooc[n][m] * mask[m]  (fp32 in LDS).
  for (int job = wave; job < 25; job += 8) {   // 5 x 5 tiles
    const int nt = job / 5, mt = job % 5;
    v8f acc = {};
#pragma unroll
    for (int kc = 0; kc < 8; ++kc) {
      v16h a = load_a_f16(Qs + nt * 16 * XS + kc * 32, XS);
      v16h bfr = load_b_f16(Ks + mt * 16 * XS + kc * 32, XS);
      acc = wmma_step(a, bfr, acc);
    }
    const int lane = tid & 31;
    const int mg = mt * 16 + (lane & 15);
    const int nb = (lane >> 4) * 8;
    const float modm = 0.0625f * Ms[mg];       // 1/sqrt(256) = 1/16
#pragma unroll
    for (int v = 0; v < 8; ++v) {
      const int ng = nt * 16 + nb + v;
      Ss[ng * SS + mg] = acc[v] * modm * cooc[ng * NCLS + mg];
    }
  }
  __syncthreads();

  // Softmax over m (row length 80), write attn as f16 padded to K=96.
  if (tid < NCLS) {
    float* row = Ss + tid * SS;
    float mx = row[0];
    for (int m = 1; m < NCLS; ++m) mx = fmaxf(mx, row[m]);
    float sum = 0.f;
    for (int m = 0; m < NCLS; ++m) { const float e = __expf(row[m] - mx); row[m] = e; sum += e; }
    const float inv = 1.f / sum;
    half_t* arow = As + tid * AS;
    for (int m = 0; m < NCLS; ++m) arow[m] = (half_t)(row[m] * inv);
    for (int m = NCLS; m < 96; ++m) arow[m] = (half_t)0.f;
  }
  __syncthreads();

  // Phase 3: O[n][h] = sum_m attn[n][m] * V[m][h]  (K padded 80->96).
  // V B-fragments for a given h-tile hoisted and reused across 5 n-tiles.
  for (int ht = wave; ht < 16; ht += 8) {
    v16h bf[3];
#pragma unroll
    for (int kc = 0; kc < 3; ++kc)
      bf[kc] = load_b_colmaj(Vs, XS, kc * 32, ht * 16, NCLS);

    const int lane = tid & 31;
    const int ncol = lane & 15;
    const int mbase = (lane >> 4) * 8;

#pragma unroll 1
    for (int nt = 0; nt < 5; ++nt) {
      v8f acc = {};
#pragma unroll
      for (int kc = 0; kc < 3; ++kc) {
        v16h a = load_a_f16(As + nt * 16 * AS + kc * 32, AS);
        acc = wmma_step(a, bf[kc], acc);
      }
#pragma unroll
      for (int v = 0; v < 8; ++v)
        Os[(nt * 16 + mbase + v) * XS + ht * 16 + ncol] = (half_t)acc[v];
    }
  }
  __syncthreads();

  // Phase 4: out[b][n][o] = O @ Wo^T + bo  (fp32 store to HBM).
  // Wo B-fragments hoisted and reused across the 5 n-tiles.
  for (int ot = wave; ot < 16; ot += 8) {
    v16h bf[8];
#pragma unroll
    for (int kc = 0; kc < 8; ++kc)
      bf[kc] = load_b_f32g(Wo + (size_t)(ot * 16) * HDIM + kc * 32, HDIM);

    const int lane = tid & 31;
    const int ncol = lane & 15;
    const int mbase = (lane >> 4) * 8;
    const float bval = bo[ot * 16 + ncol];

#pragma unroll 1
    for (int nt = 0; nt < 5; ++nt) {
      v8f acc = {};
#pragma unroll
      for (int kc = 0; kc < 8; ++kc) {
        v16h a = load_a_f16(Os + nt * 16 * XS + kc * 32, XS);
        acc = wmma_step(a, bf[kc], acc);
      }
#pragma unroll
      for (int v = 0; v < 8; ++v) {
        const int ng = nt * 16 + mbase + v;
        out[((size_t)b * NCLS + ng) * HDIM + ot * 16 + ncol] = acc[v] + bval;
      }
    }
  }
}

extern "C" void kernel_launch(void* const* d_in, const int* in_sizes, int n_in,
                              void* d_out, int out_size, void* d_ws, size_t ws_size,
                              hipStream_t stream) {
  const float* x    = (const float*)d_in[0];
  const float* Wq   = (const float*)d_in[1];
  const float* bq   = (const float*)d_in[2];
  const float* Wk   = (const float*)d_in[3];
  const float* bk   = (const float*)d_in[4];
  const float* Wv   = (const float*)d_in[5];
  const float* bv   = (const float*)d_in[6];
  const float* Wo   = (const float*)d_in[7];
  const float* bo   = (const float*)d_in[8];
  const float* cooc = (const float*)d_in[9];
  const int*   lab  = (const int*)d_in[10];
  float* out = (float*)d_out;

  const int B = in_sizes[0] / (NCLS * HDIM);   // 4096
  cooc_attn_kernel<<<B, BDIM, 0, stream>>>(x, Wq, bq, Wk, bk, Wv, bv, Wo, bo, cooc, lab, out);
}
